// TransformerEncoder_10144712753560
// MI455X (gfx1250) — matrix operand order repaired and checked by reading, not compiled, not run on hardware
//
#include <hip/hip_runtime.h>

#define DINLINE __device__ __forceinline__

#ifndef __has_builtin
#define __has_builtin(x) 0
#endif
#if __has_builtin(__builtin_amdgcn_global_load_async_to_lds_b128)
#define ASYNC_LDS 1
#else
#define ASYNC_LDS 0
#endif
#if __has_builtin(__builtin_amdgcn_s_wait_asynccnt)
#define HAVE_WAIT_ASYNC 1
#else
#define HAVE_WAIT_ASYNC 0
#endif

typedef __bf16        v16bf __attribute__((ext_vector_type(16)));
typedef float         v8f   __attribute__((ext_vector_type(8)));
typedef unsigned int  v8u   __attribute__((ext_vector_type(8)));
typedef int           v4i   __attribute__((ext_vector_type(4)));

constexpr int   L_    = 6;
constexpr int   E_    = 768;
constexpr int   FF_   = 3072;
constexpr int   HN_   = 4;
constexpr int   DH_   = 192;     // head dim
constexpr int   SEQ_  = 1024;
constexpr int   BATCH_= 8;
constexpr int   MTOK_ = BATCH_ * SEQ_;            // 8192 token rows
constexpr float SCALE_ = 0.03608439182435161f;    // 768^-0.5 (reference scales by E^-0.5)

// ---------------------------------------------------------------------------
// helpers
// ---------------------------------------------------------------------------
DINLINE unsigned short f2bf(float f) {           // round-to-nearest-even f32->bf16
  unsigned int u = __builtin_bit_cast(unsigned int, f);
  u += 0x7FFFu + ((u >> 16) & 1u);
  return (unsigned short)(u >> 16);
}

DINLINE v8f vzero() {
  v8f z;
#pragma unroll
  for (int i = 0; i < 8; ++i) z[i] = 0.f;
  return z;
}

// 16-byte global -> LDS copy; async (ASYNCcnt-tracked) when available
DINLINE void copy16(const unsigned short* g, unsigned short* l) {
#if ASYNC_LDS
  __builtin_amdgcn_global_load_async_to_lds_b128(
      (__attribute__((address_space(1))) v4i*)g,
      (__attribute__((address_space(3))) v4i*)l, 0, 0);
#else
  *(uint4*)l = *(const uint4*)g;
#endif
}

DINLINE void async_wait() {
#if ASYNC_LDS
#if HAVE_WAIT_ASYNC
  __builtin_amdgcn_s_wait_asynccnt(0);
#elif defined(__AMDGCN__)
  asm volatile("s_wait_asynccnt 0x0" ::: "memory");
#endif
#endif
}

// A-matrix fragment (16x32 bf16, lane = M, K-pair swizzle per ISA table):
// lanes 0-15 hold K {0..7,16..23}, lanes 16-31 hold K {8..15,24..31}
DINLINE v16bf load_a_frag(const unsigned short* base, int ld, int lane) {
  const int half = lane >> 4, lm = lane & 15;
  v8u t;
#pragma unroll
  for (int j = 0; j < 8; ++j) {
    int kp = (j < 4 ? j : j + 4) + half * 4;     // K-pair index
    t[j] = *(const unsigned int*)(base + lm * ld + kp * 2);
  }
  return __builtin_bit_cast(v16bf, t);
}

// B-matrix fragment from a K-major LDS tile (tile[k][n], row stride ld elts):
// lane = N (lane & 15), half-slots hold K in the same K-pair swizzle as A.
// col_base points at tile[0][n0].
DINLINE v16bf load_b_frag_t(const unsigned short* col_base, int ld, int lane) {
  const int half = lane >> 4, lm = lane & 15;
  v8u t;
#pragma unroll
  for (int j = 0; j < 8; ++j) {
    int kp = (j < 4 ? j : j + 4) + half * 4;     // K-pair index
    unsigned int lo = col_base[(kp * 2) * ld + lm];
    unsigned int hi = col_base[(kp * 2 + 1) * ld + lm];
    t[j] = lo | (hi << 16);
  }
  return __builtin_bit_cast(v16bf, t);
}

DINLINE v8f wmma_bf16(v16bf a, v16bf b, v8f c) {
  return __builtin_amdgcn_wmma_f32_16x16x32_bf16(false, a, false, b,
                                                 (short)0, c, false, false);
}

// ---------------------------------------------------------------------------
// fp32 -> bf16 bulk conversion (weights)
// ---------------------------------------------------------------------------
__global__ void cvt_kernel(const float* __restrict__ s,
                           unsigned short* __restrict__ d, long n) {
  long i = (long)blockIdx.x * blockDim.x + threadIdx.x;
  long stride = (long)gridDim.x * blockDim.x;
  for (; i < n; i += stride) d[i] = f2bf(s[i]);
}

// ---------------------------------------------------------------------------
// LayerNorm: one block per token row, fp32 in -> bf16 out
// ---------------------------------------------------------------------------
__global__ __launch_bounds__(256)
void ln_kernel(const float* __restrict__ x, const float* __restrict__ sc,
               const float* __restrict__ bi, unsigned short* __restrict__ out) {
  const int row = blockIdx.x, tid = threadIdx.x;
  const float* xr = x + (size_t)row * E_;
  float s = 0.f, s2 = 0.f;
  for (int j = tid; j < E_; j += 256) { float t = xr[j]; s += t; s2 += t * t; }
#pragma unroll
  for (int m = 1; m < 32; m <<= 1) { s += __shfl_xor(s, m); s2 += __shfl_xor(s2, m); }
  __shared__ float red[2][8];
  if ((tid & 31) == 0) { red[0][tid >> 5] = s; red[1][tid >> 5] = s2; }
  __syncthreads();
  float ts = 0.f, t2 = 0.f;
#pragma unroll
  for (int w = 0; w < 8; ++w) { ts += red[0][w]; t2 += red[1][w]; }
  const float mu  = ts * (1.0f / E_);
  const float var = t2 * (1.0f / E_) - mu * mu;
  const float inv = rsqrtf(var + 1e-5f);
  unsigned short* orow = out + (size_t)row * E_;
  for (int j = tid; j < E_; j += 256)
    orow[j] = f2bf((xr[j] - mu) * inv * sc[j] + bi[j]);
}

// ---------------------------------------------------------------------------
// Tiled WMMA GEMM: C[M,N] = A[M,K](bf16) @ W[N,K](bf16)^T  (+epilogue)
// block tile 128x128, 256 threads = 8 waves (4 along M x 2 along N),
// each wave owns 32x64 = 2x4 16x16 fragments.  A tile is async-copied
// [m][k]; W tile is stored K-major (Bst[k][n], pad 8) so B fragments are
// gathered as 16-bit elements down k (stride 136).  EPI: 0=bf16 out, 1=bias+gelu(erf)->bf16,
// 2=bias+residual->fp32
// ---------------------------------------------------------------------------
template <int EPI>
__global__ __launch_bounds__(256)
void gemm_bf16_kernel(const unsigned short* __restrict__ A,
                      const unsigned short* __restrict__ W,
                      const float* __restrict__ bias,
                      const float* __restrict__ resid,
                      unsigned short* __restrict__ outb,
                      float* __restrict__ outf,
                      int M, int N, int K) {
  __shared__ __align__(16) unsigned short As[128 * 32];
  __shared__ __align__(16) unsigned short Bst[32][136];   // [k][n], pad 8

  const int tid = threadIdx.x, lane = tid & 31, wave = tid >> 5;
  const int wm = wave & 3, wn = wave >> 2;
  const int m0 = blockIdx.y * 128, n0 = blockIdx.x * 128;

  v8f acc[2][4];
#pragma unroll
  for (int i = 0; i < 2; ++i)
#pragma unroll
    for (int j = 0; j < 4; ++j) acc[i][j] = vzero();

  for (int k0 = 0; k0 < K; k0 += 32) {
    __syncthreads();
#pragma unroll
    for (int i = 0; i < 2; ++i) {               // 512 16B chunks / 256 threads
      int c = tid + 256 * i;
      int row = c >> 2, col8 = (c & 3) * 8;
      const unsigned short* ga = A + (size_t)(m0 + row) * K + k0 + col8;
      copy16(ga, As + row * 32 + col8);         // async -> LDS, [m][k]
      const unsigned short* gb = W + (size_t)(n0 + row) * K + k0 + col8;
      uint4 dv = *(const uint4*)gb;             // transpose-scatter to [k][n]
      const unsigned short* pv = (const unsigned short*)&dv;
#pragma unroll
      for (int e = 0; e < 8; ++e) Bst[col8 + e][row] = pv[e];
      if (k0 + 32 < K) __builtin_prefetch(gb + 32, 0, 1);
    }
    async_wait();
    __syncthreads();

    v16bf af[2], bfr[4];
#pragma unroll
    for (int i = 0; i < 2; ++i)
      af[i] = load_a_frag(As + (wm * 32 + i * 16) * 32, 32, lane);
#pragma unroll
    for (int j = 0; j < 4; ++j)
      bfr[j] = load_b_frag_t(&Bst[0][wn * 64 + j * 16], 136, lane);
#pragma unroll
    for (int i = 0; i < 2; ++i)
#pragma unroll
      for (int j = 0; j < 4; ++j)
        acc[i][j] = wmma_bf16(af[i], bfr[j], acc[i][j]);
  }

  // epilogue: D layout -> VGPR r holds M=r (lanes 0-15) / M=r+8 (lanes 16-31)
  const int half = lane >> 4, lm = lane & 15;
#pragma unroll
  for (int i = 0; i < 2; ++i)
#pragma unroll
    for (int j = 0; j < 4; ++j)
#pragma unroll
      for (int r = 0; r < 8; ++r) {
        const int row = m0 + wm * 32 + i * 16 + r + half * 8;
        const int col = n0 + wn * 64 + j * 16 + lm;
        float v = acc[i][j][r];
        if (EPI != 0) v += bias[col];
        if (EPI == 0) {
          outb[(size_t)row * N + col] = f2bf(v);
        } else if (EPI == 1) {
          float g = 0.5f * v * (1.0f + erff(v * 0.70710678118654752f));
          outb[(size_t)row * N + col] = f2bf(g);
        } else {
          outf[(size_t)row * N + col] = resid[(size_t)row * N + col] + v;
        }
      }
}

// ---------------------------------------------------------------------------
// Flash-style WMMA attention.  Per the reference's reshape (no transpose),
// head hh = s>>8 and per-(b,hh) Q/K/V are contiguous [1024 x 192] matrices.
// Block: 128 threads (4 waves), 64 query rows (16 per wave), key blocks of 32.
// K is stored d-major (Kt[d][t]) for QK^T B-fragments; V stays native [t][d]
// which is already the B-fragment layout for P@V.
// ---------------------------------------------------------------------------
__global__ __launch_bounds__(128)
void attn_kernel(const unsigned short* __restrict__ q,
                 const unsigned short* __restrict__ k,
                 const unsigned short* __restrict__ v,
                 unsigned short* __restrict__ o) {
  __shared__ __align__(16) unsigned short Qs[64 * 192];   // 24 KB, [s][d]
  __shared__ __align__(16) unsigned short Kt[192][40];    // 15 KB, [d][t] pad 8
  __shared__ __align__(16) unsigned short Vs[32 * 192];   // 12 KB, [t][d]
  __shared__ __align__(16) unsigned short Ps[4][16 * 32]; //  4 KB, per-wave P

  const int tid = threadIdx.x, lane = tid & 31, wave = tid >> 5;
  const int half = lane >> 4, lm = lane & 15;
  const int bh = blockIdx.y;                       // b*4 + hh
  const size_t base = (size_t)bh * (SEQ_ * DH_);
  const int q0 = blockIdx.x * 64;

  // Q block: 64x192 bf16 = 1536 16B-chunks / 128 threads (async)
  for (int i = 0; i < 12; ++i) {
    int c = tid + 128 * i;
    int row = c / 24, ch = (c % 24) * 8;
    copy16(q + base + (size_t)(q0 + row) * 192 + ch, Qs + row * 192 + ch);
  }

  float rowmax[8], rowsum[8];
#pragma unroll
  for (int r = 0; r < 8; ++r) { rowmax[r] = -1e30f; rowsum[r] = 0.f; }
  v8f oacc[12];
#pragma unroll
  for (int n = 0; n < 12; ++n) oacc[n] = vzero();

  for (int t0 = 0; t0 < SEQ_; t0 += 32) {
    __syncthreads();                               // Kt/Vs reuse guard
    for (int i = 0; i < 6; ++i) {                  // 768 chunks / 128 threads
      int c = tid + 128 * i;
      int row = c / 24, ch = (c % 24) * 8;
      copy16(v + base + (size_t)(t0 + row) * 192 + ch, Vs + row * 192 + ch);
      uint4 dv = *(const uint4*)(k + base + (size_t)(t0 + row) * 192 + ch);
      const unsigned short* pv = (const unsigned short*)&dv;
#pragma unroll
      for (int e = 0; e < 8; ++e) Kt[ch + e][row] = pv[e];  // transpose K
    }
    async_wait();
    __syncthreads();

    // S tile = Q(16x192) @ K(32x192)^T, per wave: 2 N-frags x 6 K-steps
    v8f sf[2];
    sf[0] = vzero(); sf[1] = vzero();
#pragma unroll
    for (int kk = 0; kk < 6; ++kk) {
      v16bf aq = load_a_frag(Qs + (wave * 16) * 192 + kk * 32, 192, lane);
#pragma unroll
      for (int j = 0; j < 2; ++j) {
        v16bf bk = load_b_frag_t(&Kt[kk * 32][j * 16], 40, lane);
        sf[j] = wmma_bf16(aq, bk, sf[j]);
      }
    }

    // streaming softmax on C-layout fragments (16-lane xor reductions)
#pragma unroll
    for (int r = 0; r < 8; ++r) {
      float s0 = sf[0][r] * SCALE_;
      float s1 = sf[1][r] * SCALE_;
      float mx = fmaxf(s0, s1);
#pragma unroll
      for (int m = 1; m < 16; m <<= 1) mx = fmaxf(mx, __shfl_xor(mx, m));
      float mnew = fmaxf(rowmax[r], mx);
      float corr = __expf(rowmax[r] - mnew);
      float p0 = __expf(s0 - mnew);
      float p1 = __expf(s1 - mnew);
      float ps = p0 + p1;
#pragma unroll
      for (int m = 1; m < 16; m <<= 1) ps += __shfl_xor(ps, m);
      rowsum[r] = rowsum[r] * corr + ps;
      rowmax[r] = mnew;
#pragma unroll
      for (int n = 0; n < 12; ++n) oacc[n][r] *= corr;
      Ps[wave][(r + half * 8) * 32 + lm +  0] = f2bf(p0);
      Ps[wave][(r + half * 8) * 32 + lm + 16] = f2bf(p1);
    }
    __syncthreads();

    // O += P(16x32) @ V(32x192): 12 N-frags, single K-step; V is B-native
    v16bf ap = load_a_frag(Ps[wave], 32, lane);
#pragma unroll
    for (int n = 0; n < 12; ++n) {
      v16bf bv = load_b_frag_t(Vs + n * 16, 192, lane);
      oacc[n] = wmma_bf16(ap, bv, oacc[n]);
    }
  }

  // write O/rowsum with the reference's output remap: out[b, ss, hh*192+d]
  const int b = bh >> 2, hh = bh & 3;
#pragma unroll
  for (int n = 0; n < 12; ++n)
#pragma unroll
    for (int r = 0; r < 8; ++r) {
      const int ss = q0 + wave * 16 + r + half * 8;
      const int d  = n * 16 + lm;
      float val = oacc[n][r] / rowsum[r];
      o[((size_t)b * SEQ_ + ss) * E_ + hh * DH_ + d] = f2bf(val);
    }
}

// ---------------------------------------------------------------------------
// host orchestration
// ---------------------------------------------------------------------------
extern "C" void kernel_launch(void* const* d_in, const int* in_sizes, int n_in,
                              void* d_out, int out_size, void* d_ws, size_t ws_size,
                              hipStream_t stream) {
  (void)in_sizes; (void)n_in; (void)out_size; (void)ws_size;

  const float* hs   = (const float*)d_in[0];
  const float* ln1s = (const float*)d_in[1];
  const float* ln1b = (const float*)d_in[2];
  const float* wq   = (const float*)d_in[3];
  const float* wk   = (const float*)d_in[4];
  const float* wv   = (const float*)d_in[5];
  const float* wo   = (const float*)d_in[6];
  const float* bo   = (const float*)d_in[7];
  const float* ln2s = (const float*)d_in[8];
  const float* ln2b = (const float*)d_in[9];
  const float* w1   = (const float*)d_in[10];
  const float* b1   = (const float*)d_in[11];
  const float* w2   = (const float*)d_in[12];
  const float* b2   = (const float*)d_in[13];

  char* wsp = (char*)d_ws;
  size_t off = 0;
  auto alloc = [&](size_t bytes) -> void* {
    void* p = wsp + off;
    off = (off + bytes + 255) & ~(size_t)255;
    return p;
  };

  unsigned short* wq_b = (unsigned short*)alloc((size_t)L_ * E_ * E_ * 2);
  unsigned short* wk_b = (unsigned short*)alloc((size_t)L_ * E_ * E_ * 2);
  unsigned short* wv_b = (unsigned short*)alloc((size_t)L_ * E_ * E_ * 2);
  unsigned short* wo_b = (unsigned short*)alloc((size_t)L_ * E_ * E_ * 2);
  unsigned short* w1_b = (unsigned short*)alloc((size_t)L_ * FF_ * E_ * 2);
  unsigned short* w2_b = (unsigned short*)alloc((size_t)L_ * E_ * FF_ * 2);
  float*          x    = (float*)alloc((size_t)MTOK_ * E_ * 4);
  unsigned short* hbuf = (unsigned short*)alloc((size_t)MTOK_ * E_ * 2);
  unsigned short* qb   = (unsigned short*)alloc((size_t)MTOK_ * E_ * 2);
  unsigned short* kb   = (unsigned short*)alloc((size_t)MTOK_ * E_ * 2);
  unsigned short* vb   = (unsigned short*)alloc((size_t)MTOK_ * E_ * 2);
  unsigned short* ob   = (unsigned short*)alloc((size_t)MTOK_ * E_ * 2);
  unsigned short* fb   = (unsigned short*)alloc((size_t)MTOK_ * FF_ * 2);

  cvt_kernel<<<1024, 256, 0, stream>>>(wq, wq_b, (long)L_ * E_ * E_);
  cvt_kernel<<<1024, 256, 0, stream>>>(wk, wk_b, (long)L_ * E_ * E_);
  cvt_kernel<<<1024, 256, 0, stream>>>(wv, wv_b, (long)L_ * E_ * E_);
  cvt_kernel<<<1024, 256, 0, stream>>>(wo, wo_b, (long)L_ * E_ * E_);
  cvt_kernel<<<1024, 256, 0, stream>>>(w1, w1_b, (long)L_ * FF_ * E_);
  cvt_kernel<<<1024, 256, 0, stream>>>(w2, w2_b, (long)L_ * E_ * FF_);
  (void)hipMemcpyAsync(x, hs, (size_t)MTOK_ * E_ * 4, hipMemcpyDeviceToDevice, stream);

  const dim3 gE(E_ / 128, MTOK_ / 128);     // (6, 64)
  const dim3 gF(FF_ / 128, MTOK_ / 128);    // (24, 64)

  for (int i = 0; i < L_; ++i) {
    ln_kernel<<<MTOK_, 256, 0, stream>>>(x, ln1s + i * E_, ln1b + i * E_, hbuf);
    gemm_bf16_kernel<0><<<gE, 256, 0, stream>>>(hbuf, wq_b + (size_t)i * E_ * E_,
        nullptr, nullptr, qb, nullptr, MTOK_, E_, E_);
    gemm_bf16_kernel<0><<<gE, 256, 0, stream>>>(hbuf, wk_b + (size_t)i * E_ * E_,
        nullptr, nullptr, kb, nullptr, MTOK_, E_, E_);
    gemm_bf16_kernel<0><<<gE, 256, 0, stream>>>(hbuf, wv_b + (size_t)i * E_ * E_,
        nullptr, nullptr, vb, nullptr, MTOK_, E_, E_);
    attn_kernel<<<dim3(SEQ_ / 64, BATCH_ * HN_), 128, 0, stream>>>(qb, kb, vb, ob);
    gemm_bf16_kernel<2><<<gE, 256, 0, stream>>>(ob, wo_b + (size_t)i * E_ * E_,
        bo + i * E_, x, nullptr, x, MTOK_, E_, E_);
    ln_kernel<<<MTOK_, 256, 0, stream>>>(x, ln2s + i * E_, ln2b + i * E_, hbuf);
    gemm_bf16_kernel<1><<<gF, 256, 0, stream>>>(hbuf, w1_b + (size_t)i * FF_ * E_,
        b1 + i * FF_, nullptr, fb, nullptr, MTOK_, FF_, E_);
    gemm_bf16_kernel<2><<<gE, 256, 0, stream>>>(fb, w2_b + (size_t)i * E_ * FF_,
        b2 + i * E_, x, nullptr, x, MTOK_, E_, FF_);
  }

  (void)hipMemcpyAsync(d_out, x, (size_t)MTOK_ * E_ * 4, hipMemcpyDeviceToDevice, stream);
}
